// VectorQuantizer_85203561218632
// MI455X (gfx1250) — matrix operand order repaired and checked by reading, not compiled, not run on hardware
//
#include <hip/hip_runtime.h>
#include <hip/hip_bf16.h>

typedef __attribute__((ext_vector_type(16))) __bf16 v16bf;
typedef __attribute__((ext_vector_type(8)))  float  v8f;

#define NBATCH 32
#define CDIM   256
#define HDIM   64
#define WDIM   64
#define HW     (HDIM * WDIM)          // 4096
#define NROWS  (NBATCH * HW)          // 131072
#define NCODES 1024
#define KSTEPS (CDIM / 32)            // 8
#define NTILES (NCODES / 16)          // 64
#define TPC    2                      // code-tiles per LDS chunk
#define NCHUNK (NTILES / TPC)         // 32
#define CHUNK_BYTES (TPC * KSTEPS * 32 * 16 * 2)   // 16384 B
#define ZELEMS ((size_t)NBATCH * CDIM * HW)        // 33554432

// ---------------------------------------------------------------------------
// CDNA5 async global->LDS copy (ASYNCcnt path), per CDNA5 ISA.
// GVS addressing: mem = SADDR(64) + VADDR(32) ; LDS addr from VDST VGPR.
// ---------------------------------------------------------------------------
__device__ __forceinline__ void async_b128(unsigned lds_addr, unsigned goff,
                                           const void* sbase) {
  asm volatile("global_load_async_to_lds_b128 %0, %1, %2"
               :: "v"(lds_addr), "v"(goff), "s"(sbase) : "memory");
}
__device__ __forceinline__ void wait_async0() {
  asm volatile("s_wait_asynccnt 0" ::: "memory");
}
__device__ __forceinline__ unsigned lds_off(const void* p) {
  // Generic LDS address: low 32 bits == LDS byte offset (ISA 10.2 aperture map)
  return (unsigned)(size_t)p;
}

// ---------------------------------------------------------------------------
// Pack codebook into bf16 WMMA B-fragment order + compute ||e_k||^2.
// B order used here (contiguous k, same as A): lanes 0-15 -> K=0..15, lanes 16-31 -> K=16..31,
// element e -> K = 16*half + e, column n = lane & 15.
// ---------------------------------------------------------------------------
__global__ void __launch_bounds__(256) vq_pack_codebook(
    const float* __restrict__ ew, __bf16* __restrict__ bpack,
    float* __restrict__ normE) {
  __shared__ float tile[16][CDIM];   // 16 KB
  const int t = threadIdx.x;
  const int ct = blockIdx.x;         // code tile (16 codes)
#pragma unroll
  for (int i = 0; i < 16; ++i) {
    int li = i * 256 + t;
    int r = li >> 8, c = li & 255;   // consecutive t -> consecutive c
    tile[r][c] = ew[(size_t)(ct * 16 + r) * CDIM + c];
  }
  __syncthreads();
  if (t < 16) {
    float s = 0.f;
    for (int c = 0; c < CDIM; ++c) { float v = tile[t][c]; s += v * v; }
    normE[ct * 16 + t] = s;
  }
  const int s    = t >> 5;
  const int lane = t & 31;
  const int n    = lane & 15;
  const int half = lane >> 4;
  __bf16* dst = bpack + (((size_t)ct * KSTEPS + s) * 32 + lane) * 16;
#pragma unroll
  for (int e = 0; e < 16; ++e) {
    int k = 16 * half + e;
    dst[e] = (__bf16)tile[n][s * 32 + k];   // B[k][n] = e_n[k]
  }
}

// ---------------------------------------------------------------------------
// Main kernel: distances via bf16 WMMA, fused argmin.
// 8 waves/block; wave owns a 16-row tile. Codebook double-buffered through
// LDS with async global->LDS DMA overlapping the WMMA compute.
// ---------------------------------------------------------------------------
__global__ void __launch_bounds__(256) vq_argmin(
    const float* __restrict__ z_e, const __bf16* __restrict__ bpack,
    const float* __restrict__ normE, int* __restrict__ codes,
    float* __restrict__ codesf) {
  __shared__ __align__(32) __bf16 btile[2][TPC * KSTEPS * 32 * 16]; // 2x16KB
  __shared__ float nE[NCODES];                                      // 4 KB

  const int t    = threadIdx.x;
  const int lane = t & 31;
  const int wave = t >> 5;
  const int half = lane >> 4;
  const int m    = lane & 15;

  // Stage ||e||^2 once (visible after the first barrier below).
  for (int i = t; i < NCODES; i += 256) nE[i] = normE[i];

  // This wave's 16-row tile of z (rows share b,h; w = wbase + m).
  const int rowbase = (blockIdx.x * 8 + wave) * 16;
  const int b    = rowbase >> 12;
  const int rem  = rowbase & 4095;
  const int h    = rem >> 6;
  const int wb   = rem & 63;
  const size_t idx0 = ((size_t)b * CDIM) * HW + (size_t)h * WDIM + wb;

  // Load A fragments (contiguous k = 16*half + e, matching B), convert f32 -> bf16.
  v16bf afrag[KSTEPS];
#pragma unroll
  for (int s = 0; s < KSTEPS; ++s) {
#pragma unroll
    for (int e = 0; e < 16; ++e) {
      int k = 16 * half + e;
      int c = s * 32 + k;
      afrag[s][e] = (__bf16)z_e[idx0 + (size_t)c * HW + m];
    }
  }

  float minv[8];
  int   mini[8];
#pragma unroll
  for (int r = 0; r < 8; ++r) { minv[r] = 3.4e38f; mini[r] = 0; }

  const unsigned lb[2] = { lds_off(&btile[0][0]), lds_off(&btile[1][0]) };
  const unsigned row16 = (unsigned)t * 16u;

  // Prefetch chunk 0 into buffer 0 (4 x b128 per thread = 16 KB per block).
#pragma unroll
  for (int i = 0; i < 4; ++i)
    async_b128(lb[0] + row16 + i * 4096u, row16 + i * 4096u, bpack);

  for (int chunk = 0; chunk < NCHUNK; ++chunk) {
    wait_async0();        // own async copies for this chunk complete
    __syncthreads();      // everyone's copies done; other buffer now free

    if (chunk + 1 < NCHUNK) {
      const unsigned dst = lb[(chunk + 1) & 1];
      const unsigned src = (unsigned)(chunk + 1) * CHUNK_BYTES + row16;
#pragma unroll
      for (int i = 0; i < 4; ++i)
        async_b128(dst + row16 + i * 4096u, src + i * 4096u, bpack);
    }

    const __bf16* cur = btile[chunk & 1];
#pragma unroll
    for (int tt = 0; tt < TPC; ++tt) {
      v8f acc = {};
#pragma unroll
      for (int s = 0; s < KSTEPS; ++s) {
        const v16bf bfrag =
            *(const v16bf*)(cur + ((tt * KSTEPS + s) * 32 + lane) * 16);
        acc = __builtin_amdgcn_wmma_f32_16x16x32_bf16(
            false, afrag[s], false, bfrag, (short)0, acc, false, false);
      }
      const int code = (chunk * TPC + tt) * 16 + m;
      const float en = nE[code];
#pragma unroll
      for (int r = 0; r < 8; ++r) {
        float d = en - 2.0f * acc[r];     // ||e||^2 - 2 z.e (||z||^2 dropped)
        if (d < minv[r]) { minv[r] = d; mini[r] = code; }
      }
    }
  }

  // Reduce across the 16 lanes holding each row's columns (width-16 groups).
#pragma unroll
  for (int r = 0; r < 8; ++r) {
    float v = minv[r];
    int   i = mini[r];
#pragma unroll
    for (int mask = 8; mask; mask >>= 1) {
      float ov = __shfl_xor(v, mask, 16);
      int   oi = __shfl_xor(i, mask, 16);
      if (ov < v || (ov == v && oi < i)) { v = ov; i = oi; }
    }
    if (m == 0) {
      int row = rowbase + r + 8 * half;   // lanes 0..15 -> row r, 16..31 -> r+8
      codes[row]  = i;
      codesf[row] = (float)i;
    }
  }
}

// ---------------------------------------------------------------------------
// Gather z_q, write straight-through output (== z_q in forward), accumulate
// squared error for the loss.
// ---------------------------------------------------------------------------
__global__ void __launch_bounds__(256) vq_gather(
    const float* __restrict__ z_e, const float* __restrict__ ew,
    const int* __restrict__ codes, float* __restrict__ zq_out,
    float* __restrict__ loss_acc) {
  __shared__ float red[256];
  const int t = threadIdx.x;
  const int rowbase = blockIdx.x * 16;
  const int b   = rowbase >> 12;
  const int rem = rowbase & 4095;
  const int h   = rem >> 6;
  const int wb  = rem & 63;
  const size_t idx0 = ((size_t)b * CDIM) * HW + (size_t)h * WDIM + wb;
  const int w    = t & 15;
  const int csub = t >> 4;
  const int code = codes[rowbase + w];
  float lsum = 0.f;
#pragma unroll 4
  for (int cc = 0; cc < 16; ++cc) {
    int c = cc * 16 + csub;
    float zq = ew[(size_t)code * CDIM + c];
    size_t gi = idx0 + (size_t)c * HW + w;   // 64B-coalesced along w
    float ze = z_e[gi];
    zq_out[gi] = zq;
    float d = zq - ze;
    lsum += d * d;
  }
  red[t] = lsum;
  __syncthreads();
  for (int s = 128; s; s >>= 1) {
    if (t < s) red[t] += red[t + s];
    __syncthreads();
  }
  if (t == 0) atomicAdd(loss_acc, red[0]);
}

__global__ void vq_init(float* loss_acc) { *loss_acc = 0.f; }

__global__ void vq_finalize(const float* __restrict__ loss_acc,
                            float* __restrict__ out_loss) {
  // codebook_loss + 0.25*commitment_loss; equal values in forward
  *out_loss = 1.25f * (*loss_acc) / (float)ZELEMS;
}

// ---------------------------------------------------------------------------
extern "C" void kernel_launch(void* const* d_in, const int* in_sizes, int n_in,
                              void* d_out, int out_size, void* d_ws,
                              size_t ws_size, hipStream_t stream) {
  const float* z_e = (const float*)d_in[0];   // [32,256,64,64] f32
  const float* ew  = (const float*)d_in[1];   // [1024,256] f32

  float* out      = (float*)d_out;
  float* zq_out   = out;                  // 33,554,432 f32
  float* loss_out = out + ZELEMS;         // 1 f32
  float* codesf   = out + ZELEMS + 1;     // 131,072 f32

  char* ws = (char*)d_ws;
  __bf16* bpack   = (__bf16*)ws;                 // 524,288 B packed codebook
  float*  normE   = (float*)(ws + 524288);       //   4,096 B ||e||^2
  int*    codes   = (int*)(ws + 528384);         // 524,288 B argmin indices
  float*  loss_ac = (float*)(ws + 1052672);      //       4 B accumulator

  vq_init<<<1, 1, 0, stream>>>(loss_ac);
  vq_pack_codebook<<<NTILES, 256, 0, stream>>>(ew, bpack, normE);
  vq_argmin<<<NROWS / 128, 256, 0, stream>>>(z_e, bpack, normE, codes, codesf);
  vq_gather<<<NROWS / 16, 256, 0, stream>>>(z_e, ew, codes, zq_out, loss_ac);
  vq_finalize<<<1, 1, 0, stream>>>(loss_ac, loss_out);
}
